// CausalSelfAttention_90778428768322
// MI455X (gfx1250) — compile-verified, matrix operand order checked by reading, not run on hardware
//
#include <hip/hip_runtime.h>
#include <hip/hip_bf16.h>

// Causal self-attention: B=1, L=4096, E=1024, H=16, hd=64, fp32 in/out.
// bf16 WMMA (v_wmma_f32_16x16x32_bf16) everywhere, f32 accumulate.
// Attention: 8 waves per block share K/V tiles staged into LDS via
// GLOBAL_LOAD_ASYNC_TO_LDS_B128 (ASYNCcnt) + workgroup barriers.

#define NH   16
#define EMB  1024
#define SEQ  4096
#define HD   64

typedef __bf16 bf16_t;
typedef __attribute__((ext_vector_type(16))) __bf16 v16bf;
typedef __attribute__((ext_vector_type(8)))  __bf16 v8bf;
typedef __attribute__((ext_vector_type(4)))  __bf16 v4bf;
typedef __attribute__((ext_vector_type(8)))  float  v8f;
typedef __attribute__((ext_vector_type(4)))  int    v4i_t;

static __device__ __forceinline__ bf16_t f2bf(float f) {
  return (bf16_t)f;                      // native v_cvt (RNE)
}

static __device__ __forceinline__ v8f wmma_bf16(v16bf a, v16bf b, v8f c) {
  // D = A(16x32) * B(32x16) + C(16x16), f32 accumulate
  return __builtin_amdgcn_wmma_f32_16x16x32_bf16(false, a, false, b,
                                                 (short)0, c, false, false);
}

// Load one WMMA operand (A-style and B-style share the per-lane pattern):
// lane<16 -> K = {col0+0..7, col0+16..23}; lane>=16 -> K = {col0+8..15, col0+24..31}
static __device__ __forceinline__ v16bf load_op_bf(const bf16_t* __restrict__ base,
                                                   int row, int stride, int col0, int half) {
  const v8bf* p0 = (const v8bf*)(base + (size_t)row * stride + col0 + 8 * half);
  const v8bf* p1 = (const v8bf*)(base + (size_t)row * stride + col0 + 16 + 8 * half);
  v8bf lo = *p0, hi = *p1;
  v16bf r;
#pragma unroll
  for (int i = 0; i < 8; ++i) { r[i] = lo[i]; r[i + 8] = hi[i]; }
  return r;
}

// ---- async global -> LDS 16B copy (gfx1250 GLOBAL_LOAD_ASYNC_TO_LDS_B128) ----
#if __has_builtin(__builtin_amdgcn_global_load_async_to_lds_b128)
#define HAVE_ASYNC_LDS 1
#else
#define HAVE_ASYNC_LDS 0
#endif

static __device__ __forceinline__ void async_copy16(const bf16_t* gsrc, bf16_t* ldst) {
#if HAVE_ASYNC_LDS
  typedef __attribute__((address_space(1))) v4i_t* gvec_p;
  typedef __attribute__((address_space(3))) v4i_t* lvec_p;
  __builtin_amdgcn_global_load_async_to_lds_b128((gvec_p)gsrc, (lvec_p)ldst, 0, 0);
#else
  *(v8bf*)ldst = *(const v8bf*)gsrc;
#endif
}

static __device__ __forceinline__ void wait_async_lds() {
#if HAVE_ASYNC_LDS
#if __has_builtin(__builtin_amdgcn_s_wait_asynccnt)
  __builtin_amdgcn_s_wait_asynccnt(0);
#else
  asm volatile("s_wait_asynccnt 0x0" ::: "memory");
#endif
#endif
}

// ---------------- Kernel 0: f32 -> bf16 bulk convert (bandwidth bound) -------
__global__ __launch_bounds__(256) void cvt_kernel(const float* __restrict__ src,
                                                  bf16_t* __restrict__ dst, int n4) {
  int i = blockIdx.x * blockDim.x + threadIdx.x;
  if (i >= n4) return;
  const float4* p = (const float4*)src + i;
  float4 v = *p;
  v4bf o;
  o[0] = f2bf(v.x); o[1] = f2bf(v.y); o[2] = f2bf(v.z); o[3] = f2bf(v.w);
  *((v4bf*)dst + i) = o;
}

// ---------------- Kernel 1: qkv = x @ w_attn^T, scatter to Q/K/V^T (bf16) ----
// One wave computes a 32(m) x 64(n) tile: 8 WMMAs per 32-deep k-step.
__global__ __launch_bounds__(256) void qkv_kernel(const bf16_t* __restrict__ xb,
                                                  const bf16_t* __restrict__ wb,
                                                  bf16_t* __restrict__ Qh,
                                                  bf16_t* __restrict__ Kh,
                                                  bf16_t* __restrict__ VhT) {
  int wid  = (blockIdx.x * blockDim.x + threadIdx.x) >> 5;
  int lane = threadIdx.x & 31;
  int lq = lane & 15, half = lane >> 4;
  int mt = wid / 48;          // 128 row tiles of 32
  int nt = wid % 48;          // 48 col tiles of 64  (N = 3072)
  int m0 = mt * 32, n0 = nt * 64;

  v8f c[2][4] = {{v8f{}, v8f{}, v8f{}, v8f{}}, {v8f{}, v8f{}, v8f{}, v8f{}}};
  for (int k0 = 0; k0 < EMB; k0 += 32) {
    v16bf a0 = load_op_bf(xb, m0 + lq,      EMB, k0, half);
    v16bf a1 = load_op_bf(xb, m0 + 16 + lq, EMB, k0, half);
#pragma unroll
    for (int t = 0; t < 4; ++t) {
      v16bf b = load_op_bf(wb, n0 + t * 16 + lq, EMB, k0, half);
      c[0][t] = wmma_bf16(a0, b, c[0][t]);
      c[1][t] = wmma_bf16(a1, b, c[1][t]);
    }
  }
#pragma unroll
  for (int mh = 0; mh < 2; ++mh) {
#pragma unroll
    for (int t = 0; t < 4; ++t) {
      int n_col = n0 + t * 16 + lq;
      int which = n_col >> 10;      // 0=Q 1=K 2=V
      int e = n_col & 1023;
      int h = e >> 6;
      int d = e & 63;
#pragma unroll
      for (int r = 0; r < 8; ++r) {
        int m = m0 + mh * 16 + r + 8 * half;
        float v = c[mh][t][r];
        if (which == 0)      Qh[((size_t)(h * SEQ + m)) * HD + d] = f2bf(v * 0.125f);
        else if (which == 1) Kh[((size_t)(h * SEQ + m)) * HD + d] = f2bf(v);
        else                 VhT[((size_t)(h * HD + d)) * SEQ + m] = f2bf(v);
      }
    }
  }
}

// ---------------- Kernel 2: flash attention ----------------------------------
// One block (8 waves) = one head's 128-query stripe. K/V tiles for each 32-key
// step staged once into LDS (async DMA) and shared by all 8 waves.
__global__ __launch_bounds__(256) void attn_kernel(const bf16_t* __restrict__ Qh,
                                                   const bf16_t* __restrict__ Kh,
                                                   const bf16_t* __restrict__ VhT,
                                                   bf16_t* __restrict__ Y) {
  __shared__ bf16_t kbuf[32 * HD];   // 32 keys x 64 d   (4 KB)
  __shared__ bf16_t vbuf[HD * 32];   // 64 d  x 32 keys  (4 KB)

  int tid  = threadIdx.x;
  int wave = tid >> 5;
  int lane = tid & 31;
  int lq = lane & 15, half = lane >> 4;

  int h      = blockIdx.x >> 5;      // 16 heads
  int stripe = blockIdx.x & 31;      // 32 stripes of 128 queries
  int qbase  = stripe * 128;
  int q0     = qbase + wave * 16;

  const bf16_t* Qb = Qh + (size_t)h * SEQ * HD;
  const bf16_t* Kb = Kh + (size_t)h * SEQ * HD;
  const bf16_t* Vb = VhT + (size_t)h * HD * SEQ;

  // Q tile as two B operands (lane = query, K = head-dim), lives in regs
  v16bf qop0 = load_op_bf(Qb, q0 + lq, HD, 0, half);
  v16bf qop1 = load_op_bf(Qb, q0 + lq, HD, 32, half);

  v8f o[4] = {v8f{}, v8f{}, v8f{}, v8f{}};
  float m_run = -1e30f, l_run = 0.0f;
  const int query = q0 + lq;

  // cooperative fill: each thread moves one 16B chunk per buffer
  int eo   = tid * 8;                // bf16 element offset, 0..2040
  int kRow = eo >> 6, kCol = eo & 63;   // K tile: 32 x 64
  int vRow = eo >> 5, vCol = eo & 31;   // V^T tile: 64 x 32

  for (int j0 = 0; j0 <= qbase + 127; j0 += 32) {
    async_copy16(Kb + (size_t)(j0 + kRow) * HD + kCol, &kbuf[eo]);
    async_copy16(Vb + (size_t)vRow * SEQ + j0 + vCol,  &vbuf[eo]);
    wait_async_lds();
    __syncthreads();                 // LDS tiles visible to all waves

    if (j0 <= q0 + 15) {             // wave-uniform: EXEC stays all-ones
      // S^T tiles: lane = query, VGPR r -> key = j0 + (tile*16) + r + 8*half
      v8f s0 = v8f{}, s1 = v8f{};
      {
        v16bf ka = load_op_bf(kbuf, lq,      HD, 0,  half);
        v16bf kb = load_op_bf(kbuf, lq,      HD, 32, half);
        s0 = wmma_bf16(ka, qop0, s0);
        s0 = wmma_bf16(kb, qop1, s0);
        v16bf kc = load_op_bf(kbuf, 16 + lq, HD, 0,  half);
        v16bf kd = load_op_bf(kbuf, 16 + lq, HD, 32, half);
        s1 = wmma_bf16(kc, qop0, s1);
        s1 = wmma_bf16(kd, qop1, s1);
      }

      bool need_mask = (j0 + 31 > q0);
      float sv0[8], sv1[8];
#pragma unroll
      for (int r = 0; r < 8; ++r) {
        int k0i = j0 + r + 8 * half;
        sv0[r] = (need_mask && (k0i > query))      ? -1e30f : s0[r];
        sv1[r] = (need_mask && (k0i + 16 > query)) ? -1e30f : s1[r];
      }

      // online softmax: keys for a row live in lanes q and q+16 only
      float mloc = sv0[0];
#pragma unroll
      for (int r = 0; r < 8; ++r) { mloc = fmaxf(mloc, sv0[r]); mloc = fmaxf(mloc, sv1[r]); }
      mloc = fmaxf(mloc, __shfl_xor(mloc, 16, 32));
      float m_new = fmaxf(m_run, mloc);
      float alpha = __expf(m_run - m_new);

      float psum = 0.0f;
      v16bf pa;                      // P repacks directly into A-operand layout
#pragma unroll
      for (int r = 0; r < 8; ++r) {
        float p0 = __expf(sv0[r] - m_new);
        float p1 = __expf(sv1[r] - m_new);
        psum += p0 + p1;
        pa[r]     = f2bf(p0);
        pa[r + 8] = f2bf(p1);
      }
      psum += __shfl_xor(psum, 16, 32);
      l_run = alpha * l_run + psum;
      m_run = m_new;

      // rescale O rows (O row = r + 8*half; alpha lives at lane == row index)
#pragma unroll
      for (int r = 0; r < 8; ++r) {
        float ar = __shfl(alpha, r + 8 * half, 32);
        o[0][r] *= ar; o[1][r] *= ar; o[2][r] *= ar; o[3][r] *= ar;
      }

      // O += P * V  (B operand from V^T rows in LDS: contiguous in key)
#pragma unroll
      for (int t = 0; t < 4; ++t) {
        v16bf vb = load_op_bf(vbuf, t * 16 + lq, 32, 0, half);
        o[t] = wmma_bf16(pa, vb, o[t]);
      }
    }
    __syncthreads();                 // compute done before tiles overwritten
  }

  // normalize and store Y[q][h*64+d] (bf16)
#pragma unroll
  for (int r = 0; r < 8; ++r) {
    float lr  = __shfl(l_run, r + 8 * half, 32);
    float inv = (lr > 0.0f) ? (1.0f / lr) : 0.0f;
    int qrow = q0 + r + 8 * half;
#pragma unroll
    for (int t = 0; t < 4; ++t) {
      Y[(size_t)qrow * EMB + h * HD + t * 16 + lq] = f2bf(o[t][r] * inv);
    }
  }
}

// ---------------- Kernel 3: out = Y @ w_proj^T (f32 out), 32x64 per wave -----
__global__ __launch_bounds__(256) void proj_kernel(const bf16_t* __restrict__ Y,
                                                   const bf16_t* __restrict__ wb,
                                                   float* __restrict__ out) {
  int wid  = (blockIdx.x * blockDim.x + threadIdx.x) >> 5;
  int lane = threadIdx.x & 31;
  int lq = lane & 15, half = lane >> 4;
  int mt = wid >> 4;          // 128 row tiles of 32
  int nt = wid & 15;          // 16 col tiles of 64
  int m0 = mt * 32, n0 = nt * 64;

  v8f c[2][4] = {{v8f{}, v8f{}, v8f{}, v8f{}}, {v8f{}, v8f{}, v8f{}, v8f{}}};
  for (int k0 = 0; k0 < EMB; k0 += 32) {
    v16bf a0 = load_op_bf(Y, m0 + lq,      EMB, k0, half);
    v16bf a1 = load_op_bf(Y, m0 + 16 + lq, EMB, k0, half);
#pragma unroll
    for (int t = 0; t < 4; ++t) {
      v16bf b = load_op_bf(wb, n0 + t * 16 + lq, EMB, k0, half);
      c[0][t] = wmma_bf16(a0, b, c[0][t]);
      c[1][t] = wmma_bf16(a1, b, c[1][t]);
    }
  }
#pragma unroll
  for (int mh = 0; mh < 2; ++mh) {
#pragma unroll
    for (int t = 0; t < 4; ++t) {
      int n = n0 + t * 16 + lq;
#pragma unroll
      for (int r = 0; r < 8; ++r) {
        int m = m0 + mh * 16 + r + 8 * half;
        out[(size_t)m * EMB + n] = c[mh][t][r];
      }
    }
  }
}

extern "C" void kernel_launch(void* const* d_in, const int* in_sizes, int n_in,
                              void* d_out, int out_size, void* d_ws, size_t ws_size,
                              hipStream_t stream) {
  (void)in_sizes; (void)n_in; (void)out_size; (void)ws_size;
  const float* x      = (const float*)d_in[0];
  const float* w_attn = (const float*)d_in[1];
  const float* w_proj = (const float*)d_in[2];
  float* out = (float*)d_out;

  char* ws = (char*)d_ws;
  size_t off = 0;
  bf16_t* xb  = (bf16_t*)(ws + off); off += (size_t)SEQ * EMB * 2;        // 8 MB
  bf16_t* wab = (bf16_t*)(ws + off); off += (size_t)3 * EMB * EMB * 2;    // 6 MB
  bf16_t* wpb = (bf16_t*)(ws + off); off += (size_t)EMB * EMB * 2;        // 2 MB
  bf16_t* Qh  = (bf16_t*)(ws + off); off += (size_t)NH * SEQ * HD * 2;    // 8 MB
  bf16_t* Kh  = (bf16_t*)(ws + off); off += (size_t)NH * SEQ * HD * 2;    // 8 MB
  bf16_t* VhT = (bf16_t*)(ws + off); off += (size_t)NH * SEQ * HD * 2;    // 8 MB
  bf16_t* Yb  = (bf16_t*)(ws + off); off += (size_t)SEQ * EMB * 2;        // 8 MB

  // bulk f32 -> bf16 conversions (x, w_attn, w_proj)
  {
    int n4;
    n4 = SEQ * EMB / 4;
    cvt_kernel<<<(n4 + 255) / 256, 256, 0, stream>>>(x, xb, n4);
    n4 = 3 * EMB * EMB / 4;
    cvt_kernel<<<(n4 + 255) / 256, 256, 0, stream>>>(w_attn, wab, n4);
    n4 = EMB * EMB / 4;
    cvt_kernel<<<(n4 + 255) / 256, 256, 0, stream>>>(w_proj, wpb, n4);
  }

  // 128 m-tiles * 48 n64-tiles = 6144 waves, 8 waves/block
  qkv_kernel<<<6144 / 8, 256, 0, stream>>>(xb, wab, Qh, Kh, VhT);
  // 16 heads * 32 query-stripes = 512 blocks
  attn_kernel<<<NH * 32, 256, 0, stream>>>(Qh, Kh, VhT, Yb);
  // 128 m-tiles * 16 n64-tiles = 2048 waves
  proj_kernel<<<2048 / 8, 256, 0, stream>>>(Yb, wpb, out);
}
